// CGCNN_MultiTask_51771535786263
// MI455X (gfx1250) — compile-verified, matrix operand order checked by reading, not run on hardware
//
#include <hip/hip_runtime.h>
#include <stdint.h>
#include <stddef.h>

#define N_NODES 50000
#define N_EDGES 800000
#define N_GRAPHS 256
#define NODE_IN 92
#define EDGE_IN 41
#define H 128
#define N_CONV 3
#define N_HID 2
#define N_PROP 3

typedef __bf16 bf16_t;
typedef __attribute__((ext_vector_type(16))) __bf16 v16bf;
typedef __attribute__((ext_vector_type(8)))  __bf16 v8bf;
typedef __attribute__((ext_vector_type(8)))  float  v8f;

// ---------- helpers ----------
__device__ inline bf16_t f2bf(float f) {
  union { float f; uint32_t u; } x; x.f = f;
  uint32_t r = x.u + 0x7FFFu + ((x.u >> 16) & 1u);
  union { uint16_t u; bf16_t b; } y; y.u = (uint16_t)(r >> 16);
  return y.b;
}
__device__ inline float softplusf(float x) {
  return fmaxf(x, 0.0f) + __logf(1.0f + __expf(-fabsf(x)));
}
__device__ inline void atomic_add_f32(float* p, float v) {
  __hip_atomic_fetch_add(p, v, __ATOMIC_RELAXED, __HIP_MEMORY_SCOPE_AGENT);
}
__device__ inline v8f zero8() {
  v8f z = {0.f,0.f,0.f,0.f,0.f,0.f,0.f,0.f};
  return z;
}
// Fragment loader matching the ISA 16-bit 16x32 A layout.
// p points at (row, kBase + laneHi8); second half is at +16 halfwords.
__device__ inline v16bf ld_frag(const bf16_t* p) {
  union { v16bf v; v8bf h[2]; } u;
  u.h[0] = *reinterpret_cast<const v8bf*>(p);
  u.h[1] = *reinterpret_cast<const v8bf*>(p + 16);
  return u.v;
}

// ---------- CDNA5 async global->LDS copy (ASYNCcnt path), guarded ----------
#if defined(__has_builtin)
#  if __has_builtin(__builtin_amdgcn_global_load_async_to_lds_b128)
#    define HAVE_ASYNC_LDS 1
#  endif
#endif

// The builtin's pointer params are 16-byte int-vector pointers in AS1/AS3
// (clang diagnostic: "__vector_size__(4*sizeof(int)) int __device__ *").
typedef int gv4i __attribute__((vector_size(16)));
typedef __attribute__((address_space(1))) gv4i* gptr_v4i;
typedef __attribute__((address_space(3))) gv4i* lptr_v4i;

__device__ inline void async_copy16(const bf16_t* g, bf16_t* l) {
#ifdef HAVE_ASYNC_LDS
  __builtin_amdgcn_global_load_async_to_lds_b128(
      (gptr_v4i)g, (lptr_v4i)l, 0, 0);
#else
  *(v8bf*)l = *(const v8bf*)g;
#endif
}
__device__ inline void async_wait0() {
#ifdef HAVE_ASYNC_LDS
#  if __has_builtin(__builtin_amdgcn_s_wait_asynccnt)
  __builtin_amdgcn_s_wait_asynccnt(0);
#  else
  asm volatile("s_wait_asynccnt 0" ::: "memory");
#  endif
#endif
}

// LDS strides (bf16 elements), padded so (stride dwords) % 64 != 0
#define S384 392   // 196 dwords
#define S256 264   // 132 dwords

// ---------- weight transpose+bf16 prep ----------
__global__ void prep_weights_kernel(const float* __restrict__ e1W,
                                    const float* __restrict__ e2W,
                                    const float* __restrict__ nW,
                                    bf16_t* __restrict__ W1t,
                                    bf16_t* __restrict__ W2t,
                                    bf16_t* __restrict__ Wnt) {
  const int R1 = N_CONV * 256 * 384;   // W1t: [l][o=256][k=384]
  const int R2 = N_CONV * 128 * 256;   // W2t/Wnt: [l][o=128][k=256]
  int stride = gridDim.x * blockDim.x;
  for (int i = blockIdx.x * blockDim.x + threadIdx.x; i < R1 + 2 * R2; i += stride) {
    if (i < R1) {
      int l = i / (256 * 384), r = i % (256 * 384);
      int o = r / 384, k = r % 384;
      W1t[i] = f2bf(e1W[(size_t)l * 384 * 256 + (size_t)k * 256 + o]);
    } else if (i < R1 + R2) {
      int j = i - R1;
      int l = j / (128 * 256), r = j % (128 * 256);
      int o = r / 256, k = r % 256;
      W2t[j] = f2bf(e2W[(size_t)l * 256 * 128 + (size_t)k * 128 + o]);
    } else {
      int j = i - R1 - R2;
      int l = j / (128 * 256), r = j % (128 * 256);
      int o = r / 256, k = r % 256;
      Wnt[j] = f2bf(nW[(size_t)l * 256 * 128 + (size_t)k * 128 + o]);
    }
  }
}

// ---------- zero ----------
__global__ void zero_kernel(float* __restrict__ p, size_t n) {
  size_t i = (size_t)blockIdx.x * blockDim.x + threadIdx.x;
  size_t st = (size_t)gridDim.x * blockDim.x;
  for (; i < n; i += st) p[i] = 0.0f;
}

// ---------- input embeddings (small K, VALU) ----------
__global__ void node_embed_kernel(const float* __restrict__ x,
                                  const float* __restrict__ W,
                                  const float* __restrict__ b,
                                  float* __restrict__ h,
                                  bf16_t* __restrict__ hbf) {
  __shared__ float sW[NODE_IN * H];
  __shared__ float sb[H];
  for (int i = threadIdx.x; i < NODE_IN * H; i += blockDim.x) sW[i] = W[i];
  if (threadIdx.x < H) sb[threadIdx.x] = b[threadIdx.x];
  __syncthreads();
  int c = threadIdx.x;  // blockDim = 128
  for (int n = blockIdx.x; n < N_NODES; n += gridDim.x) {
    const float* xr = x + (size_t)n * NODE_IN;
    float s = sb[c];
    for (int k = 0; k < NODE_IN; ++k) s = fmaf(xr[k], sW[k * H + c], s);
    h[(size_t)n * H + c] = s;
    hbf[(size_t)n * H + c] = f2bf(s);
  }
}

__global__ void edge_embed_kernel(const float* __restrict__ ea,
                                  const float* __restrict__ W,
                                  const float* __restrict__ b,
                                  bf16_t* __restrict__ eabf) {
  __shared__ float sW[EDGE_IN * H];
  __shared__ float sb[H];
  for (int i = threadIdx.x; i < EDGE_IN * H; i += blockDim.x) sW[i] = W[i];
  if (threadIdx.x < H) sb[threadIdx.x] = b[threadIdx.x];
  __syncthreads();
  int c = threadIdx.x;  // blockDim = 128
  for (int e = blockIdx.x; e < N_EDGES; e += gridDim.x) {
    const float* er = ea + (size_t)e * EDGE_IN;
    float s = sb[c];
    for (int k = 0; k < EDGE_IN; ++k) s = fmaf(er[k], sW[k * H + c], s);
    eabf[(size_t)e * H + c] = f2bf(s);
  }
}

// ---------- edge MLP + scatter (the big WMMA kernel) ----------
// LDS: W1s 256x392 bf16 (200704B) | W2s 128x264 bf16 (67584B) |
//      As  64x392 bf16 (50176B, overlaid by Es 64x264) | cix 64 ints (256B)
// total 318720B <= 320KB WGP LDS
__global__ __launch_bounds__(256, 1) void edge_mlp_kernel(
    const bf16_t* __restrict__ hbf, const bf16_t* __restrict__ eabf,
    const bf16_t* __restrict__ W1t, const bf16_t* __restrict__ W2t,
    const float* __restrict__ b1, const float* __restrict__ b2,
    const int* __restrict__ rowIdx, const int* __restrict__ colIdx,
    float* __restrict__ aggr) {
  extern __shared__ __align__(16) char smem[];
  bf16_t* W1s = (bf16_t*)smem;                                   // 256 x S384
  bf16_t* W2s = (bf16_t*)(smem + 256 * S384 * 2);                // 128 x S256
  bf16_t* As  = (bf16_t*)(smem + (256 * S384 + 128 * S256) * 2); // 64 x S384
  int* cix    = (int*)(smem + (256 * S384 + 128 * S256 + 64 * S384) * 2);

  const int tid = threadIdx.x;
  // stage transposed weights into LDS (padded rows) via async global->LDS
  for (int i = tid; i < 256 * 48; i += 256) {
    int r = i / 48, c8 = (i % 48) * 8;
    async_copy16(W1t + (size_t)r * 384 + c8, W1s + r * S384 + c8);
  }
  for (int i = tid; i < 128 * 32; i += 256) {
    int r = i / 32, c8 = (i % 32) * 8;
    async_copy16(W2t + (size_t)r * 256 + c8, W2s + r * S256 + c8);
  }
  async_wait0();
  __syncthreads();

  const int lane = tid & 31;
  const int wave = tid >> 5;
  const int wm = wave & 3;        // M subtile (16 edges)
  const int wn = wave >> 2;       // N half
  const int lrow = lane & 15;
  const int lhi8 = (lane >> 4) << 3;

  const int nTiles = N_EDGES / 64;
  for (int tile = blockIdx.x; tile < nTiles; tile += gridDim.x) {
    const int e0 = tile * 64;
    {  // gather: 4 threads per edge, each copies a 32-col quarter of 3 segments
      int el = tid >> 2, q = tid & 3;
      int e = e0 + el;
      int rsrc = rowIdx[e], csrc = colIdx[e];
      if (q == 0) cix[el] = csrc;
      const bf16_t* hr = hbf + (size_t)rsrc * H + q * 32;
      const bf16_t* hc = hbf + (size_t)csrc * H + q * 32;
      const bf16_t* ee = eabf + (size_t)e * H + q * 32;
      bf16_t* d = As + el * S384 + q * 32;
      // prefetch next tile's edge features while we compute this one
      int en = e + 64 * gridDim.x;
      if (en < N_EDGES) __builtin_prefetch(eabf + (size_t)en * H, 0, 1);
#pragma unroll
      for (int j = 0; j < 4; ++j) {
        async_copy16(hr + j * 8, d + j * 8);
        async_copy16(hc + j * 8, d + 128 + j * 8);
        async_copy16(ee + j * 8, d + 256 + j * 8);
      }
      async_wait0();
    }
    __syncthreads();

    // GEMM1: wave computes 16 edges x 128 cols (8 N-tiles), K=384
    v8f acc[8];
#pragma unroll
    for (int t = 0; t < 8; ++t) acc[t] = zero8();
    const bf16_t* arow = As + (wm * 16 + lrow) * S384 + lhi8;
#pragma unroll
    for (int kb = 0; kb < 384; kb += 32) {
      v16bf a = ld_frag(arow + kb);
      v16bf bfr[8];
#pragma unroll
      for (int t = 0; t < 8; ++t)
        bfr[t] = ld_frag(W1s + (wn * 128 + t * 16 + lrow) * S384 + kb + lhi8);
#pragma unroll
      for (int t = 0; t < 8; ++t)
        acc[t] = __builtin_amdgcn_wmma_f32_16x16x32_bf16(
            false, a, false, bfr[t], (short)0, acc[t], false, false);
    }
    __syncthreads();  // all As reads complete before overlay

    // bias + softplus -> Es (bf16, overlays As)
    bf16_t* Es = As;
#pragma unroll
    for (int t = 0; t < 8; ++t) {
      int c = wn * 128 + t * 16 + lrow;
      float bias = b1[c];
#pragma unroll
      for (int r = 0; r < 8; ++r) {
        int m = wm * 16 + lhi8 + r;  // C layout: lanes>=16 hold M=r+8
        Es[m * S256 + c] = f2bf(softplusf(acc[t][r] + bias));
      }
    }
    __syncthreads();

    // GEMM2: wave computes 16 edges x 64 cols (4 N-tiles), K=256
    v8f acc2[4];
#pragma unroll
    for (int t = 0; t < 4; ++t) acc2[t] = zero8();
    const bf16_t* arow2 = Es + (wm * 16 + lrow) * S256 + lhi8;
#pragma unroll
    for (int kb = 0; kb < 256; kb += 32) {
      v16bf a = ld_frag(arow2 + kb);
      v16bf bfr[4];
#pragma unroll
      for (int t = 0; t < 4; ++t)
        bfr[t] = ld_frag(W2s + (wn * 64 + t * 16 + lrow) * S256 + kb + lhi8);
#pragma unroll
      for (int t = 0; t < 4; ++t)
        acc2[t] = __builtin_amdgcn_wmma_f32_16x16x32_bf16(
            false, a, false, bfr[t], (short)0, acc2[t], false, false);
    }

    // bias + softplus + scatter-add into aggr[col[e]]
    int dsts[8];
#pragma unroll
    for (int r = 0; r < 8; ++r) dsts[r] = cix[wm * 16 + lhi8 + r];
#pragma unroll
    for (int t = 0; t < 4; ++t) {
      int c = wn * 64 + t * 16 + lrow;
      float bias = b2[c];
#pragma unroll
      for (int r = 0; r < 8; ++r) {
        float v = softplusf(acc2[t][r] + bias);
        atomic_add_f32(aggr + (size_t)dsts[r] * H + c, v);
      }
    }
    __syncthreads();  // before next tile rewrites As/cix
  }
}

// ---------- node update (WMMA, K=256 = [h | aggr]) ----------
// LDS: Wns 128x264 bf16 (67584B) | Ans 64x264 bf16 (33792B) -> 101376B
__global__ __launch_bounds__(256, 1) void node_update_kernel(
    const bf16_t* __restrict__ Wnt, const float* __restrict__ bn,
    const float* __restrict__ aggr, float* __restrict__ h,
    bf16_t* __restrict__ hbf) {
  extern __shared__ __align__(16) char smem[];
  bf16_t* Ws = (bf16_t*)smem;                     // 128 x S256
  bf16_t* Ans = (bf16_t*)(smem + 128 * S256 * 2); // 64 x S256

  const int tid = threadIdx.x;
  for (int i = tid; i < 128 * 32; i += 256) {
    int r = i / 32, c8 = (i % 32) * 8;
    async_copy16(Wnt + (size_t)r * 256 + c8, Ws + r * S256 + c8);
  }
  async_wait0();
  __syncthreads();

  const int lane = tid & 31;
  const int wave = tid >> 5;
  const int wm = wave & 3, wn = wave >> 2;
  const int lrow = lane & 15;
  const int lhi8 = (lane >> 4) << 3;

  const int nTiles = (N_NODES + 63) / 64;
  for (int tile = blockIdx.x; tile < nTiles; tile += gridDim.x) {
    const int n0 = tile * 64;
    {  // gather [h_bf | bf16(aggr)] rows
      int nl = tid >> 2, q = tid & 3;
      int n = n0 + nl;
      bf16_t* d = Ans + nl * S256 + q * 32;
      if (n < N_NODES) {
        const bf16_t* hr = hbf + (size_t)n * H + q * 32;
        const float* ar = aggr + (size_t)n * H + q * 32;
#pragma unroll
        for (int j = 0; j < 4; ++j) async_copy16(hr + j * 8, d + j * 8);
#pragma unroll
        for (int j = 0; j < 32; ++j) d[128 + j] = f2bf(ar[j]);
      } else {
#pragma unroll
        for (int j = 0; j < 32; ++j) { d[j] = f2bf(0.f); d[128 + j] = f2bf(0.f); }
      }
      async_wait0();
    }
    __syncthreads();

    v8f acc[4];
#pragma unroll
    for (int t = 0; t < 4; ++t) acc[t] = zero8();
    const bf16_t* arow = Ans + (wm * 16 + lrow) * S256 + lhi8;
#pragma unroll
    for (int kb = 0; kb < 256; kb += 32) {
      v16bf a = ld_frag(arow + kb);
      v16bf bfr[4];
#pragma unroll
      for (int t = 0; t < 4; ++t)
        bfr[t] = ld_frag(Ws + (wn * 64 + t * 16 + lrow) * S256 + kb + lhi8);
#pragma unroll
      for (int t = 0; t < 4; ++t)
        acc[t] = __builtin_amdgcn_wmma_f32_16x16x32_bf16(
            false, a, false, bfr[t], (short)0, acc[t], false, false);
    }

    // bias + softplus + residual; write h (f32) and h_bf
#pragma unroll
    for (int t = 0; t < 4; ++t) {
      int c = wn * 64 + t * 16 + lrow;
      float bias = bn[c];
#pragma unroll
      for (int r = 0; r < 8; ++r) {
        int m = wm * 16 + lhi8 + r;
        int n = n0 + m;
        if (n < N_NODES) {
          float v = softplusf(acc[t][r] + bias) + h[(size_t)n * H + c];
          h[(size_t)n * H + c] = v;
          hbf[(size_t)n * H + c] = f2bf(v);
        }
      }
    }
    __syncthreads();
  }
}

// ---------- global mean pool ----------
__global__ void pool_kernel(const float* __restrict__ h, const int* __restrict__ batch,
                            float* __restrict__ sums, float* __restrict__ counts) {
  int c = threadIdx.x;  // blockDim = 128
  for (int n = blockIdx.x; n < N_NODES; n += gridDim.x) {
    int b = batch[n];
    atomic_add_f32(&sums[(size_t)b * H + c], h[(size_t)n * H + c]);
    if (c == 0) atomic_add_f32(&counts[b], 1.0f);
  }
}

// ---------- shared MLP + task heads (256 graphs, tiny) ----------
__global__ void head_kernel(const float* __restrict__ sums, const float* __restrict__ counts,
                            const float* __restrict__ sW1, const float* __restrict__ sb1,
                            const float* __restrict__ sW2, const float* __restrict__ sb2,
                            const float* __restrict__ hW, const float* __restrict__ hb,
                            const float* __restrict__ hoW, const float* __restrict__ hob,
                            float* __restrict__ out) {
  __shared__ float bufA[H], bufB[H], red[H];
  int g = blockIdx.x, c = threadIdx.x;  // blockDim = 128
  float cnt = fmaxf(counts[g], 1.0f);
  bufA[c] = sums[(size_t)g * H + c] / cnt;
  __syncthreads();
  float s = sb1[c];
  for (int k = 0; k < H; ++k) s = fmaf(bufA[k], sW1[k * H + c], s);
  bufB[c] = softplusf(s);
  __syncthreads();
  s = sb2[c];
  for (int k = 0; k < H; ++k) s = fmaf(bufB[k], sW2[k * H + c], s);
  float v2 = softplusf(s);
  __syncthreads();
  bufA[c] = v2;  // bufA = pooled+shared features g
  __syncthreads();
  for (int p = 0; p < N_PROP; ++p) {
    bufB[c] = bufA[c];
    __syncthreads();
    for (int l = 0; l < N_HID; ++l) {
      const float* W = hW + ((size_t)(p * N_HID + l)) * H * H;
      float t = hb[(p * N_HID + l) * H + c];
      for (int k = 0; k < H; ++k) t = fmaf(bufB[k], W[k * H + c], t);
      float v = softplusf(t);
      __syncthreads();
      bufB[c] = v;
      __syncthreads();
    }
    red[c] = bufB[c] * hoW[p * H + c];
    __syncthreads();
    if (c == 0) {
      float acc = hob[p];
      for (int k = 0; k < H; ++k) acc += red[k];
      out[p * N_GRAPHS + g] = acc;
    }
    __syncthreads();
  }
}

// ---------- launch ----------
extern "C" void kernel_launch(void* const* d_in, const int* in_sizes, int n_in,
                              void* d_out, int out_size, void* d_ws, size_t ws_size,
                              hipStream_t stream) {
  (void)in_sizes; (void)n_in; (void)out_size; (void)ws_size;
  const float* x          = (const float*)d_in[0];
  const float* edge_attr  = (const float*)d_in[1];
  const float* node_W     = (const float*)d_in[2];
  const float* node_b     = (const float*)d_in[3];
  const float* edge_W     = (const float*)d_in[4];
  const float* edge_b     = (const float*)d_in[5];
  const float* conv_e1_W  = (const float*)d_in[6];
  const float* conv_e1_b  = (const float*)d_in[7];
  const float* conv_e2_W  = (const float*)d_in[8];
  const float* conv_e2_b  = (const float*)d_in[9];
  const float* conv_n_W   = (const float*)d_in[10];
  const float* conv_n_b   = (const float*)d_in[11];
  const float* shared_W1  = (const float*)d_in[12];
  const float* shared_b1  = (const float*)d_in[13];
  const float* shared_W2  = (const float*)d_in[14];
  const float* shared_b2  = (const float*)d_in[15];
  const float* head_W     = (const float*)d_in[16];
  const float* head_b     = (const float*)d_in[17];
  const float* head_out_W = (const float*)d_in[18];
  const float* head_out_b = (const float*)d_in[19];
  const int* edge_index   = (const int*)d_in[20];
  const int* batch        = (const int*)d_in[21];
  const int* rowIdx = edge_index;
  const int* colIdx = edge_index + N_EDGES;

  char* ws = (char*)d_ws;
  size_t off = 0;
  auto alloc = [&](size_t bytes) -> void* {
    void* p = ws + off;
    off += (bytes + 255) & ~(size_t)255;
    return p;
  };
  float*  h      = (float*)alloc((size_t)N_NODES * H * 4);
  bf16_t* hbf    = (bf16_t*)alloc((size_t)N_NODES * H * 2);
  bf16_t* eabf   = (bf16_t*)alloc((size_t)N_EDGES * H * 2);
  float*  aggr   = (float*)alloc((size_t)N_NODES * H * 4);
  bf16_t* W1t    = (bf16_t*)alloc((size_t)N_CONV * 256 * 384 * 2);
  bf16_t* W2t    = (bf16_t*)alloc((size_t)N_CONV * 128 * 256 * 2);
  bf16_t* Wnt    = (bf16_t*)alloc((size_t)N_CONV * 128 * 256 * 2);
  float*  sums   = (float*)alloc((size_t)N_GRAPHS * H * 4);
  float*  counts = (float*)alloc((size_t)N_GRAPHS * 4);

  prep_weights_kernel<<<512, 256, 0, stream>>>(conv_e1_W, conv_e2_W, conv_n_W,
                                               W1t, W2t, Wnt);
  node_embed_kernel<<<2048, 128, 0, stream>>>(x, node_W, node_b, h, hbf);
  edge_embed_kernel<<<4096, 128, 0, stream>>>(edge_attr, edge_W, edge_b, eabf);

  const size_t edgeLds = (size_t)(256 * S384 + 128 * S256 + 64 * S384) * 2 + 64 * 4;
  const size_t nodeLds = (size_t)(128 * S256 + 64 * S256) * 2;
  for (int l = 0; l < N_CONV; ++l) {
    zero_kernel<<<1024, 256, 0, stream>>>(aggr, (size_t)N_NODES * H);
    edge_mlp_kernel<<<640, 256, edgeLds, stream>>>(
        hbf, eabf,
        W1t + (size_t)l * 256 * 384, W2t + (size_t)l * 128 * 256,
        conv_e1_b + l * 256, conv_e2_b + l * 128,
        rowIdx, colIdx, aggr);
    node_update_kernel<<<640, 256, nodeLds, stream>>>(
        Wnt + (size_t)l * 128 * 256, conv_n_b + l * 128, aggr, h, hbf);
  }

  zero_kernel<<<64, 256, 0, stream>>>(sums, (size_t)N_GRAPHS * H);
  zero_kernel<<<1, 256, 0, stream>>>(counts, (size_t)N_GRAPHS);
  pool_kernel<<<2048, 128, 0, stream>>>(h, batch, sums, counts);
  head_kernel<<<N_GRAPHS, 128, 0, stream>>>(sums, counts, shared_W1, shared_b1,
                                            shared_W2, shared_b2, head_W, head_b,
                                            head_out_W, head_out_b, (float*)d_out);
}
